// MBLocks_66305705116317
// MI455X (gfx1250) — matrix operand order repaired and checked by reading, not compiled, not run on hardware
//
#include <hip/hip_runtime.h>

typedef _Float16 f16;
typedef __attribute__((ext_vector_type(16))) _Float16 v16h;
typedef __attribute__((ext_vector_type(8)))  _Float16 v8h;
typedef __attribute__((ext_vector_type(8)))  float    v8f;

constexpr int NN = 20000;
constexpr int EE = 320000;
constexpr int DD = 256;       // N_HID == IN_DIM
constexpr int TT = 8;         // node types
constexpr int RR = 8;         // relations
constexpr int HH = 8;         // heads
constexpr int LL = 2;         // layers
constexpr int DK = 32;        // head dim

__device__ __forceinline__ int f2ord(float f) {
  int i = __float_as_int(f);
  return (i >= 0) ? i : (i ^ 0x7fffffff);
}
__device__ __forceinline__ float ord2f(int i) {
  return __int_as_float((i >= 0) ? i : (i ^ 0x7fffffff));
}
__device__ __forceinline__ int imin(int a, int b) { return a < b ? a : b; }

// Build the per-lane 16-half A fragment from two statically 16B-aligned chunks.
// Lane L: row = L&15, chunk offsets 2*kb and 2*kb+8 (kb = (L>>4)*8), i.e. contiguous
// k = 16*(L>>4) + s within the 32-wide K slice.
__device__ __forceinline__ v16h load_a_frag(const f16* p, int kb) {
  v8h lo = *(const v8h*)(p + 2 * kb);
  v8h hi = *(const v8h*)(p + 2 * kb + 8);
  return __builtin_shufflevector(lo, hi, 0, 1, 2, 3, 4, 5, 6, 7,
                                 8, 9, 10, 11, 12, 13, 14, 15);
}
__device__ __forceinline__ v16h zero_a_frag() {
  v16h a;
#pragma unroll
  for (int j = 0; j < 16; ++j) a[j] = (f16)0.f;
  return a;
}

// ---------------------------------------------------------------- utilities
__global__ void k_cvt_f16(const float* __restrict__ in, f16* __restrict__ out, long n) {
  long i = blockIdx.x * (long)blockDim.x + threadIdx.x;
  long st = (long)gridDim.x * blockDim.x;
  for (; i < n; i += st) out[i] = (f16)in[i];
}

// in: [B][Kd][Od] f32  ->  out: [B][Od][Kd] f16  (so K is contiguous per out-col)
__global__ void k_cvt_transpose(const float* __restrict__ in, f16* __restrict__ out,
                                int B, int Kd, int Od) {
  long total = (long)B * Kd * Od;
  long i = blockIdx.x * (long)blockDim.x + threadIdx.x;
  long st = (long)gridDim.x * blockDim.x;
  for (; i < total; i += st) {
    int o = (int)(i % Od);
    long r = i / Od;
    int k = (int)(r % Kd);
    int b = (int)(r / Kd);
    out[((long)b * Od + o) * Kd + k] = (f16)in[i];
  }
}

__global__ void k_fill_f32(float* __restrict__ p, float v, long n) {
  long i = blockIdx.x * (long)blockDim.x + threadIdx.x;
  long st = (long)gridDim.x * blockDim.x;
  for (; i < n; i += st) p[i] = v;
}
__global__ void k_fill_i32(int* __restrict__ p, int v, long n) {
  long i = blockIdx.x * (long)blockDim.x + threadIdx.x;
  long st = (long)gridDim.x * blockDim.x;
  for (; i < n; i += st) p[i] = v;
}

// counting-sort bucket build (single block). Per-element outputs downstream are
// independent of intra-bucket order, so atomic ordering does not affect results.
__global__ void k_bucket(const int* __restrict__ labels, int n, int nb,
                         int* __restrict__ off, int* __restrict__ perm) {
  __shared__ int cnt[16], cur[16];
  int tid = threadIdx.x;
  if (tid < nb) cnt[tid] = 0;
  __syncthreads();
  for (int i = tid; i < n; i += blockDim.x) atomicAdd(&cnt[labels[i]], 1);
  __syncthreads();
  if (tid == 0) {
    int acc = 0;
    for (int b = 0; b < nb; ++b) { off[b] = acc; cur[b] = acc; acc += cnt[b]; }
    off[nb] = acc;
  }
  __syncthreads();
  for (int i = tid; i < n; i += blockDim.x) {
    int p = atomicAdd(&cur[labels[i]], 1);
    perm[p] = i;
  }
}

// ------------------------------------------------- type-dispatched linear GEMM
// One wave per block. Tile: 16 rows (same node type) x 64 out cols (4 WMMA tiles).
// xin [N][256] f16 ; Wt [T][256 out][256 k] f16 ; bias [T][256] f32
__global__ void __launch_bounds__(32)
k_type_linear(const f16* __restrict__ xin, const f16* __restrict__ Wt,
              const float* __restrict__ bias, const int* __restrict__ perm,
              const int* __restrict__ toff, float* __restrict__ yf32,
              f16* __restrict__ yf16, int act /*0=none,1=tanh*/) {
  __shared__ int rowIdx[16];
  int tile = blockIdx.x;
  int t = -1, start = 0, cnt = 0, acc = 0;
  for (int ty = 0; ty < TT; ++ty) {
    int s = toff[ty], e = toff[ty + 1];
    int nt = (e - s + 15) >> 4;
    if (tile < acc + nt) {
      t = ty;
      start = s + (tile - acc) * 16;
      cnt = imin(16, e - start);
      break;
    }
    acc += nt;
  }
  if (t < 0) return;

  int lane = threadIdx.x;
  int n = lane & 15;
  int hi = lane >> 4;
  if (lane < 16) rowIdx[lane] = (lane < cnt) ? perm[start + lane] : -1;
  __syncthreads();

  int og = blockIdx.y;  // 0..3 -> 64 out cols
  v8f acc4[4];
#pragma unroll
  for (int s4 = 0; s4 < 4; ++s4)
#pragma unroll
    for (int j = 0; j < 8; ++j) acc4[s4][j] = 0.f;

  const f16* wbase = Wt + (size_t)t * DD * DD;
  int arow = rowIdx[n];
  const f16* ap = (arow >= 0) ? (xin + (size_t)arow * DD) : nullptr;
  int kb = hi * 8;

  for (int kk = 0; kk < DD; kk += 32) {
    if (kk + 32 < DD) {  // prefetch next K tile (-> global_prefetch_b8)
      if (ap) __builtin_prefetch(ap + kk + 32, 0, 3);
      __builtin_prefetch(wbase + (size_t)(og * 64 + n) * DD + kk + 32, 0, 3);
    }
    v16h a = ap ? load_a_frag(ap + kk, kb) : zero_a_frag();
#pragma unroll
    for (int s4 = 0; s4 < 4; ++s4) {
      int o = og * 64 + s4 * 16 + n;
      v16h b = *(const v16h*)(wbase + (size_t)o * DD + kk + hi * 16);
      acc4[s4] = __builtin_amdgcn_wmma_f32_16x16x32_f16(false, a, false, b,
                                                        (short)0, acc4[s4], false, false);
    }
  }
  // epilogue: lane holds col n, rows m = hi*8 + v
#pragma unroll
  for (int s4 = 0; s4 < 4; ++s4) {
    int o = og * 64 + s4 * 16 + n;
    float bv = bias[t * DD + o];
#pragma unroll
    for (int v = 0; v < 8; ++v) {
      int m = hi * 8 + v;
      int row = rowIdx[m];
      if (row < 0) continue;
      float val = acc4[s4][v] + bv;
      if (act == 1) val = tanhf(val);
      if (yf32) yf32[(size_t)row * DD + o] = val;
      if (yf16) yf16[(size_t)row * DD + o] = (f16)val;
    }
  }
}

// ------------------------------------------ edge pass 1: scores + segment max
// One wave per block, 16 edges (same relation) per tile, loop over 8 heads.
// k_e = k[src,h] (16x32) @ rel_att[r,h] (32x32) via two 16x16x32 WMMAs.
__global__ void __launch_bounds__(32)
k_edge_score(const f16* __restrict__ kf, const f16* __restrict__ qf,
             const f16* __restrict__ relT /*[R][H][32e][32d]*/,
             const float* __restrict__ rel_pri /*[R][H]*/,
             const int* __restrict__ eperm, const int* __restrict__ roff,
             const int* __restrict__ ei, const float* __restrict__ ew,
             float* __restrict__ scores, int* __restrict__ smax) {
  __shared__ int lep[16], lsrc[16], ldst[16];
  __shared__ float ke[16][33];
  int tile = blockIdx.x;
  int r = -1, start = 0, cnt = 0, acc = 0;
  for (int rr = 0; rr < RR; ++rr) {
    int s = roff[rr], e = roff[rr + 1];
    int nt = (e - s + 15) >> 4;
    if (tile < acc + nt) {
      r = rr;
      start = s + (tile - acc) * 16;
      cnt = imin(16, e - start);
      break;
    }
    acc += nt;
  }
  if (r < 0) return;

  int lane = threadIdx.x, n = lane & 15, hi = lane >> 4;
  if (lane < 16) {
    int ep = (lane < cnt) ? eperm[start + lane] : -1;
    lep[lane] = ep;
    lsrc[lane] = (ep >= 0) ? ei[ep] : -1;
    ldst[lane] = (ep >= 0) ? ei[EE + ep] : -1;
  }
  __syncthreads();
  const float invs = 0.17677669529663687f;  // 1/sqrt(32)
  int kb = hi * 8;
  int srow = lsrc[n];
  const f16* ap = (srow >= 0) ? (kf + (size_t)srow * DD) : nullptr;

  for (int h = 0; h < HH; ++h) {
    v16h a = ap ? load_a_frag(ap + h * DK, kb) : zero_a_frag();
    const f16* rb = relT + ((size_t)r * HH + h) * (DK * DK);
    v8f c[2];
#pragma unroll
    for (int s2 = 0; s2 < 2; ++s2) {
      v16h b = *(const v16h*)(rb + (size_t)(s2 * 16 + n) * DK + hi * 16);
      v8f cz;
#pragma unroll
      for (int j = 0; j < 8; ++j) cz[j] = 0.f;
      c[s2] = __builtin_amdgcn_wmma_f32_16x16x32_f16(false, a, false, b,
                                                     (short)0, cz, false, false);
    }
#pragma unroll
    for (int s2 = 0; s2 < 2; ++s2)
#pragma unroll
      for (int v = 0; v < 8; ++v) ke[hi * 8 + v][s2 * 16 + n] = c[s2][v];
    __syncthreads();
    if (lane < cnt) {
      const v8h* qp = (const v8h*)(qf + (size_t)ldst[lane] * DD + h * DK);
      float s = 0.f;
#pragma unroll
      for (int c4 = 0; c4 < 4; ++c4) {
        v8h qv = qp[c4];
#pragma unroll
        for (int j = 0; j < 8; ++j) s += (float)qv[j] * ke[lane][c4 * 8 + j];
      }
      s = s * rel_pri[r * HH + h] * invs * ew[lep[lane]];
      scores[(size_t)lep[lane] * HH + h] = s;
      atomicMax(&smax[ldst[lane] * HH + h], f2ord(s));
    }
    __syncthreads();
  }
}

__global__ void k_edge_expsum(const float* __restrict__ scores, const int* __restrict__ ei,
                              const int* __restrict__ smax, float* __restrict__ ssum) {
  long total = (long)EE * HH;
  long i = blockIdx.x * (long)blockDim.x + threadIdx.x;
  long st = (long)gridDim.x * blockDim.x;
  for (; i < total; i += st) {
    int e = (int)(i / HH), h = (int)(i % HH);
    int dn = ei[EE + e];
    float m = ord2f(smax[dn * HH + h]);
    atomicAdd(&ssum[dn * HH + h], __expf(scores[i] - m));
  }
}

// ------------------------------- edge pass 2: recompute v_e, scatter att*v_e
__global__ void __launch_bounds__(32)
k_edge_agg(const f16* __restrict__ vf, const f16* __restrict__ relT,
           const int* __restrict__ eperm, const int* __restrict__ roff,
           const int* __restrict__ ei, const float* __restrict__ scores,
           const int* __restrict__ smax, const float* __restrict__ ssum,
           float* __restrict__ agg) {
  __shared__ int lep[16], lsrc[16], ldst[16];
  __shared__ float latt[16];
  int tile = blockIdx.x;
  int r = -1, start = 0, cnt = 0, acc = 0;
  for (int rr = 0; rr < RR; ++rr) {
    int s = roff[rr], e = roff[rr + 1];
    int nt = (e - s + 15) >> 4;
    if (tile < acc + nt) {
      r = rr;
      start = s + (tile - acc) * 16;
      cnt = imin(16, e - start);
      break;
    }
    acc += nt;
  }
  if (r < 0) return;

  int lane = threadIdx.x, n = lane & 15, hi = lane >> 4;
  if (lane < 16) {
    int ep = (lane < cnt) ? eperm[start + lane] : -1;
    lep[lane] = ep;
    lsrc[lane] = (ep >= 0) ? ei[ep] : -1;
    ldst[lane] = (ep >= 0) ? ei[EE + ep] : -1;
  }
  __syncthreads();
  int kb = hi * 8;
  int srow = lsrc[n];
  const f16* ap = (srow >= 0) ? (vf + (size_t)srow * DD) : nullptr;

  for (int h = 0; h < HH; ++h) {
    if (lane < 16) {
      float a_ = 0.f;
      if (lane < cnt) {
        int ep = lep[lane], dn = ldst[lane];
        float s = scores[(size_t)ep * HH + h];
        float m = ord2f(smax[dn * HH + h]);
        a_ = __expf(s - m) / (ssum[dn * HH + h] + 1e-9f);
      }
      latt[lane] = a_;
    }
    __syncthreads();
    v16h a = ap ? load_a_frag(ap + h * DK, kb) : zero_a_frag();
    const f16* rb = relT + ((size_t)r * HH + h) * (DK * DK);
#pragma unroll
    for (int s2 = 0; s2 < 2; ++s2) {
      v16h b = *(const v16h*)(rb + (size_t)(s2 * 16 + n) * DK + hi * 16);
      v8f cz;
#pragma unroll
      for (int j = 0; j < 8; ++j) cz[j] = 0.f;
      v8f c = __builtin_amdgcn_wmma_f32_16x16x32_f16(false, a, false, b,
                                                     (short)0, cz, false, false);
#pragma unroll
      for (int v = 0; v < 8; ++v) {
        int m = hi * 8 + v;
        int dn = ldst[m];
        if (dn >= 0)
          atomicAdd(&agg[(size_t)dn * DD + h * DK + s2 * 16 + n], latt[m] * c[v]);
      }
    }
    __syncthreads();
  }
}

__global__ void k_gelu_cvt(const float* __restrict__ in, f16* __restrict__ out, long n) {
  long i = blockIdx.x * (long)blockDim.x + threadIdx.x;
  long st = (long)gridDim.x * blockDim.x;
  for (; i < n; i += st) {
    float x = in[i];
    float g = 0.5f * x * (1.f + tanhf(0.7978845608028654f * (x + 0.044715f * x * x * x)));
    out[i] = (f16)g;
  }
}

__global__ void __launch_bounds__(256)
k_residual_ln(const float* __restrict__ trans, const float* __restrict__ xin,
              const int* __restrict__ ntype, const float* __restrict__ skip /*[T]*/,
              const float* __restrict__ lng, const float* __restrict__ lnb /*[T][256]*/,
              float* __restrict__ xout, f16* __restrict__ xh) {
  __shared__ float red[256];
  int node = blockIdx.x, i = threadIdx.x;
  int t = ntype[node];
  float alpha = 1.f / (1.f + __expf(-skip[t]));
  size_t idx = (size_t)node * DD + i;
  float val = trans[idx] * alpha + xin[idx] * (1.f - alpha);
  red[i] = val;
  __syncthreads();
  for (int s = 128; s > 0; s >>= 1) {
    if (i < s) red[i] += red[i + s];
    __syncthreads();
  }
  float mu = red[0] / (float)DD;
  __syncthreads();
  float d = val - mu;
  red[i] = d * d;
  __syncthreads();
  for (int s = 128; s > 0; s >>= 1) {
    if (i < s) red[i] += red[i + s];
    __syncthreads();
  }
  float var = red[0] / (float)DD;
  float o = d * rsqrtf(var + 1e-5f) * lng[t * DD + i] + lnb[t * DD + i];
  xout[idx] = o;
  if (xh) xh[idx] = (f16)o;
}

__global__ void k_att_out(const float* __restrict__ scores, const int* __restrict__ ei,
                          const int* __restrict__ smax, const float* __restrict__ ssum,
                          float* __restrict__ out) {
  long total = (long)EE * HH;
  long i = blockIdx.x * (long)blockDim.x + threadIdx.x;
  long st = (long)gridDim.x * blockDim.x;
  for (; i < total; i += st) {
    int e = (int)(i / HH), h = (int)(i % HH);
    int dn = ei[EE + e];
    float m = ord2f(smax[dn * HH + h]);
    out[i] = __expf(scores[i] - m) / (ssum[dn * HH + h] + 1e-9f);
  }
}

// --------------------------------------------------------------------- launch
extern "C" void kernel_launch(void* const* d_in, const int* in_sizes, int n_in,
                              void* d_out, int out_size, void* d_ws, size_t ws_size,
                              hipStream_t stream) {
  (void)in_sizes; (void)n_in; (void)out_size; (void)ws_size;
  const float* node_feature = (const float*)d_in[0];
  const int*   node_type    = (const int*)d_in[1];
  const float* edge_weight  = (const float*)d_in[2];
  const int*   edge_index   = (const int*)d_in[3];
  const int*   edge_type    = (const int*)d_in[4];
  const float* adapt_W = (const float*)d_in[5];
  const float* adapt_b = (const float*)d_in[6];
  const float* k_W = (const float*)d_in[7];
  const float* k_b = (const float*)d_in[8];
  const float* q_W = (const float*)d_in[9];
  const float* q_b = (const float*)d_in[10];
  const float* v_W = (const float*)d_in[11];
  const float* v_b = (const float*)d_in[12];
  const float* a_W = (const float*)d_in[13];
  const float* a_b = (const float*)d_in[14];
  const float* rel_pri = (const float*)d_in[15];
  const float* rel_att = (const float*)d_in[16];
  const float* rel_msg = (const float*)d_in[17];
  const float* skip = (const float*)d_in[18];
  const float* ln_g = (const float*)d_in[19];
  const float* ln_b = (const float*)d_in[20];

  char* ws = (char*)d_ws;
  size_t off = 0;
  auto alloc = [&](size_t bytes) -> void* {
    off = (off + 255) & ~(size_t)255;
    void* p = ws + off;
    off += bytes;
    return p;
  };
  int*   perm    = (int*)alloc((size_t)NN * 4);
  int*   toff    = (int*)alloc((TT + 1) * 4);
  int*   eperm   = (int*)alloc((size_t)EE * 4);
  int*   roff    = (int*)alloc((RR + 1) * 4);
  f16*   nf16    = (f16*)alloc((size_t)NN * DD * 2);
  f16*   adaptWt = (f16*)alloc((size_t)TT * DD * DD * 2);
  f16*   kWt     = (f16*)alloc((size_t)LL * TT * DD * DD * 2);
  f16*   qWt     = (f16*)alloc((size_t)LL * TT * DD * DD * 2);
  f16*   vWt     = (f16*)alloc((size_t)LL * TT * DD * DD * 2);
  f16*   aWt     = (f16*)alloc((size_t)LL * TT * DD * DD * 2);
  f16*   relAttT = (f16*)alloc((size_t)LL * RR * HH * DK * DK * 2);
  f16*   relMsgT = (f16*)alloc((size_t)LL * RR * HH * DK * DK * 2);
  float* x       = (float*)alloc((size_t)NN * DD * 4);
  f16*   xh      = (f16*)alloc((size_t)NN * DD * 2);
  f16*   kf      = (f16*)alloc((size_t)NN * DD * 2);
  f16*   qf      = (f16*)alloc((size_t)NN * DD * 2);
  f16*   vf      = (f16*)alloc((size_t)NN * DD * 2);
  float* agg     = (float*)alloc((size_t)NN * DD * 4);
  f16*   gin     = (f16*)alloc((size_t)NN * DD * 2);
  float* trans   = (float*)alloc((size_t)NN * DD * 4);
  float* scores  = (float*)alloc((size_t)EE * HH * 4);
  int*   smax    = (int*)alloc((size_t)NN * HH * 4);
  float* ssum    = (float*)alloc((size_t)NN * HH * 4);

  // precision conversions + weight transposes (contiguous-K for WMMA B operand)
  k_cvt_f16<<<2048, 256, 0, stream>>>(node_feature, nf16, (long)NN * DD);
  k_cvt_transpose<<<2048, 256, 0, stream>>>(adapt_W, adaptWt, TT, DD, DD);
  k_cvt_transpose<<<2048, 256, 0, stream>>>(k_W, kWt, LL * TT, DD, DD);
  k_cvt_transpose<<<2048, 256, 0, stream>>>(q_W, qWt, LL * TT, DD, DD);
  k_cvt_transpose<<<2048, 256, 0, stream>>>(v_W, vWt, LL * TT, DD, DD);
  k_cvt_transpose<<<2048, 256, 0, stream>>>(a_W, aWt, LL * TT, DD, DD);
  k_cvt_transpose<<<512, 256, 0, stream>>>(rel_att, relAttT, LL * RR * HH, DK, DK);
  k_cvt_transpose<<<512, 256, 0, stream>>>(rel_msg, relMsgT, LL * RR * HH, DK, DK);

  k_bucket<<<1, 1024, 0, stream>>>(node_type, NN, TT, toff, perm);
  k_bucket<<<1, 1024, 0, stream>>>(edge_type, EE, RR, roff, eperm);

  const int NTILES = NN / 16 + TT;   // upper bound on sum of per-type tiles
  const int ETILES = EE / 16 + RR;   // upper bound on sum of per-relation tiles

  // adapt: x = tanh(type_linear(node_feature))
  k_type_linear<<<dim3(NTILES, 4), 32, 0, stream>>>(nf16, adaptWt, adapt_b, perm, toff,
                                                    x, xh, 1);

  for (int l = 0; l < LL; ++l) {
    size_t wofs = (size_t)l * TT * DD * DD;
    size_t rofs = (size_t)l * RR * HH * DK * DK;
    k_type_linear<<<dim3(NTILES, 4), 32, 0, stream>>>(xh, kWt + wofs, k_b + l * TT * DD,
                                                      perm, toff, nullptr, kf, 0);
    k_type_linear<<<dim3(NTILES, 4), 32, 0, stream>>>(xh, qWt + wofs, q_b + l * TT * DD,
                                                      perm, toff, nullptr, qf, 0);
    k_type_linear<<<dim3(NTILES, 4), 32, 0, stream>>>(xh, vWt + wofs, v_b + l * TT * DD,
                                                      perm, toff, nullptr, vf, 0);
    k_fill_i32<<<256, 256, 0, stream>>>(smax, (int)0x80000000, (long)NN * HH);
    k_fill_f32<<<256, 256, 0, stream>>>(ssum, 0.f, (long)NN * HH);
    k_fill_f32<<<2048, 256, 0, stream>>>(agg, 0.f, (long)NN * DD);

    k_edge_score<<<ETILES, 32, 0, stream>>>(kf, qf, relAttT + rofs, rel_pri + l * RR * HH,
                                            eperm, roff, edge_index, edge_weight,
                                            scores, smax);
    k_edge_expsum<<<2048, 256, 0, stream>>>(scores, edge_index, smax, ssum);
    k_edge_agg<<<ETILES, 32, 0, stream>>>(vf, relMsgT + rofs, eperm, roff, edge_index,
                                          scores, smax, ssum, agg);

    k_gelu_cvt<<<2048, 256, 0, stream>>>(agg, gin, (long)NN * DD);
    k_type_linear<<<dim3(NTILES, 4), 32, 0, stream>>>(gin, aWt + wofs, a_b + l * TT * DD,
                                                      perm, toff, trans, nullptr, 0);

    float* xout  = (l == LL - 1) ? (float*)d_out : x;
    f16*   xhout = (l == LL - 1) ? nullptr : xh;
    k_residual_ln<<<NN, 256, 0, stream>>>(trans, x, node_type, skip + l * TT,
                                          ln_g + (size_t)l * TT * DD,
                                          ln_b + (size_t)l * TT * DD, xout, xhout);
  }
  k_att_out<<<2048, 256, 0, stream>>>(scores, edge_index, smax, ssum,
                                      (float*)d_out + (size_t)NN * DD);
}
